// SegAttnModel_25769803776432
// MI455X (gfx1250) — matrix operand order repaired and checked by reading, not compiled, not run on hardware
//
#include <hip/hip_runtime.h>
#include <hip/hip_bf16.h>
#include <cstdint>
#include <cstddef>

// ---------------------------------------------------------------------------
// CDNA5 (gfx1250) wave32 WMMA types & helpers
// ---------------------------------------------------------------------------
typedef _Float16 h16v __attribute__((ext_vector_type(16)));
typedef _Float16 h8v  __attribute__((ext_vector_type(8)));
typedef _Float16 h4v  __attribute__((ext_vector_type(4)));
typedef float    f8v  __attribute__((ext_vector_type(8)));
typedef unsigned int u32x4 __attribute__((ext_vector_type(4)));
typedef int      i32x4 __attribute__((ext_vector_type(4)));
typedef int      i32x8 __attribute__((ext_vector_type(8)));

#ifdef __has_builtin
#if __has_builtin(__builtin_amdgcn_tensor_load_to_lds)
#define HAVE_TDM 1
#endif
#endif

#define DEVI __device__ __forceinline__

DEVI f8v wmma16(h16v a, h16v b, f8v c) {
  // D(16x16,f32) = A(16x32,f16) * B(32x16,f16) + C
  return __builtin_amdgcn_wmma_f32_16x16x32_f16(
      /*neg_a=*/false, a, /*neg_b=*/false, b,
      /*c_mod=*/(short)0, c, /*reuse_a=*/false, /*reuse_b=*/false);
}

// Defeat LICM: make a uniform pointer loop-variant so B-fragment loads are not
// hoisted out of the time loop (which forced a 128-VGPR scratch spill).
template <typename T>
DEVI const T* launder_ptr(const T* p) {
  asm volatile("" : "+s"(p));
  return p;
}

// A-fragment (16x32 f16) from a row-major [rows x 512] f16 buffer (LDS).
// Per ISA: lanes 0-15 hold K 0..7 & 16..23 ; lanes 16-31 hold K 8..15 & 24..31.
DEVI h16v afrag_shared(const _Float16* base, int row, int kb) {
  h16v r;
  const h8v lo = *(const h8v*)(base + row * 512 + kb);
  const h8v hi = *(const h8v*)(base + row * 512 + kb + 16);
#pragma unroll
  for (int j = 0; j < 8; ++j) { r[j] = lo[j]; r[8 + j] = hi[j]; }
  return r;
}

DEVI float sigmoidf_(float x) { return 1.f / (1.f + __expf(-x)); }

// ---------------------------------------------------------------------------
// Model constants
// ---------------------------------------------------------------------------
#define BATCH 32
#define ALPB 80
#define EMB 256
#define HS 512
#define XS 256
#define TLEN 98

// ---------------------------------------------------------------------------
// BN prep: scale = g*rsqrt(v+eps), bias = bt + (conv_b - m)*scale
// ---------------------------------------------------------------------------
__global__ void k_bnprep(const float* __restrict__ g, const float* __restrict__ bt,
                         const float* __restrict__ m, const float* __restrict__ v,
                         const float* __restrict__ convb,
                         float* __restrict__ scale, float* __restrict__ bias, int C) {
  int i = blockIdx.x * blockDim.x + threadIdx.x;
  if (i >= C) return;
  float s = g[i] * rsqrtf(v[i] + 1e-5f);
  scale[i] = s;
  bias[i] = bt[i] + (convb[i] - m[i]) * s;
}

__global__ void k_addbias(const float* __restrict__ a, const float* __restrict__ b,
                          float* __restrict__ o, int n) {
  int i = blockIdx.x * blockDim.x + threadIdx.x;
  if (i < n) o[i] = a[i] + b[i];
}

// ---------------------------------------------------------------------------
// Weight packing into WMMA B-fragment layout.
// K ordering for conv im2col: k = (kh*ksz + kw)*Cin + ci (channel innermost).
// Fragment: tile(kt,nt): lane l stores 16 halves, k = kt*32 + (l<16?0:8) + (j&7) + (j<8?0:16),
// n = nt*16 + (l&15).  out[((kt*Ntiles+nt)*32 + l)*16 + j]
// ---------------------------------------------------------------------------
__global__ void k_packW(const float* __restrict__ w, _Float16* __restrict__ out,
                        int Cout, int Cin, int ksz, int Ktiles) {
  int Ntiles = Cout >> 4;
  size_t total = (size_t)Ktiles * Ntiles * 512;
  size_t t = (size_t)blockIdx.x * blockDim.x + threadIdx.x;
  if (t >= total) return;
  int j = (int)(t & 15);
  int lane = (int)((t >> 4) & 31);
  size_t fi = t >> 9;
  int nt = (int)(fi % Ntiles);
  int kt = (int)(fi / Ntiles);
  int n = nt * 16 + (lane & 15);
  int k = kt * 32 + ((lane & 16) ? 8 : 0) + (j & 7) + ((j & 8) ? 16 : 0);
  int K = Cin * ksz * ksz;
  float v = 0.f;
  if (k < K) {
    int s = k / Cin, ci = k - s * Cin;
    int kh = s / ksz, kw = s - kh * ksz;
    v = w[(((size_t)n * Cin + ci) * ksz + kh) * ksz + kw];
  }
  out[t] = (_Float16)v;
}

// LSTM: combined K=1024 (xt 512 | h 512), N=2048; Ktiles=32, Ntiles=128.
__global__ void k_packW_lstm(const float* __restrict__ wih, const float* __restrict__ whh,
                             _Float16* __restrict__ out) {
  size_t total = (size_t)32 * 128 * 512;
  size_t t = (size_t)blockIdx.x * blockDim.x + threadIdx.x;
  if (t >= total) return;
  int j = (int)(t & 15);
  int lane = (int)((t >> 4) & 31);
  size_t fi = t >> 9;
  int nt = (int)(fi % 128);
  int kt = (int)(fi / 128);
  int n = nt * 16 + (lane & 15);
  int k = kt * 32 + ((lane & 16) ? 8 : 0) + (j & 7) + ((j & 8) ? 16 : 0);
  float v = (k < 512) ? wih[(size_t)n * 512 + k] : whh[(size_t)n * 512 + (k - 512)];
  out[t] = (_Float16)v;
}

// ---------------------------------------------------------------------------
// conv1 (Cin=1, 5x5, pad 2) + BN + ReLU + 2x2 maxpool, fused (VALU; 5% of FLOPs)
// x: (32,64,1024) f32 -> out NHWC f16 (32,32,512,32)
// ---------------------------------------------------------------------------
__global__ void k_conv1_pool(const float* __restrict__ x, const float* __restrict__ w,
                             const float* __restrict__ scale, const float* __restrict__ bias,
                             _Float16* __restrict__ out) {
  int idx = blockIdx.x * blockDim.x + threadIdx.x;
  if (idx >= 32 * 32 * 512 * 32) return;
  int co = idx & 31;
  int wo = (idx >> 5) & 511;
  int ho = (idx >> 14) & 31;
  int b  = idx >> 19;
  float sc = scale[co], bi = bias[co];
  float mx = -3.4e38f;
#pragma unroll
  for (int dy = 0; dy < 2; ++dy) {
#pragma unroll
    for (int dx = 0; dx < 2; ++dx) {
      int hc = ho * 2 + dy, wc = wo * 2 + dx;
      float acc = 0.f;
#pragma unroll
      for (int kh = 0; kh < 5; ++kh) {
        int hi = hc + kh - 2;
        if (hi < 0 || hi >= 64) continue;
#pragma unroll
        for (int kw = 0; kw < 5; ++kw) {
          int wi = wc + kw - 2;
          if (wi < 0 || wi >= 1024) continue;
          acc += x[((size_t)b * 64 + hi) * 1024 + wi] * w[co * 25 + kh * 5 + kw];
        }
      }
      float v = fmaxf(acc * sc + bi, 0.f);
      mx = fmaxf(mx, v);
    }
  }
  out[idx] = (_Float16)mx;
}

// ---------------------------------------------------------------------------
// Implicit-GEMM conv via WMMA, fully compile-time geometry.
//  act: NHWC f16 (B,HIN,WIN,CIN), zero-padded implicitly.
//  M = B*HC*WC rows (m = (b*HC+ho)*WC+wo), N = COUT, K = CIN*KSZ*KSZ.
//  Block = 128 thr (4 waves); each wave owns one 16-row M tile and a 64-col N
//  slab (4 D fragments, A reused 4x). K loop fully unrolled: all im2col
//  div/mod folds to constants (CIN>=32 pow2, WC/HC pow2).
// ---------------------------------------------------------------------------
template <int HIN, int WIN, int CIN, int KSZ, int PAD,
          int HC, int WC, int COUT, int KTILES, int RELU>
__global__ __launch_bounds__(128) void k_conv_gemm_t(
    const _Float16* __restrict__ act, const h16v* __restrict__ Bp,
    const float* __restrict__ scale, const float* __restrict__ bias,
    _Float16* __restrict__ outH, float* __restrict__ outF) {
  constexpr int NT = COUT / 16;
  const int lane = threadIdx.x & 31;
  const int wid = threadIdx.x >> 5;
  const int m0 = (blockIdx.x * 4 + wid) * 16;
  const int mrow = m0 + (lane & 15);
  const int wo = mrow % WC;              // pow2 -> and
  const int ho = (mrow / WC) % HC;       // pow2 -> shift/and
  const int b  = mrow / (WC * HC);
  const int nt0 = blockIdx.y * 4;
  const int kbofs = (lane & 16) ? 8 : 0;

  f8v acc0 = {}, acc1 = {}, acc2 = {}, acc3 = {};
#pragma unroll
  for (int kt = 0; kt < KTILES; ++kt) {
    h16v afrag;
#pragma unroll
    for (int ch = 0; ch < 2; ++ch) {
      const int kkb = kt * 32 + ch * 16;       // unroll-constant
      const int s   = kkb / CIN;               // constant: kbofs(<8) never carries
      const int cib = kkb % CIN;               // constant
      const int dh  = s / KSZ - PAD;           // constant
      const int dw  = s % KSZ - PAD;           // constant
      const int hi = ho + dh, wi = wo + dw;
      h8v v = {};
      if ((unsigned)hi < (unsigned)HIN && (unsigned)wi < (unsigned)WIN)
        v = *(const h8v*)(act + (((size_t)b * HIN + hi) * WIN + wi) * CIN + cib + kbofs);
#pragma unroll
      for (int j = 0; j < 8; ++j) afrag[ch * 8 + j] = v[j];
    }
    const h16v* bb = Bp + ((size_t)kt * NT + nt0) * 32 + lane;
    if (kt + 1 < KTILES)
      __builtin_prefetch(Bp + ((size_t)(kt + 1) * NT + nt0) * 32 + lane, 0, 0);
    h16v b0 = bb[0], b1 = bb[32], b2 = bb[64], b3 = bb[96];
    acc0 = wmma16(afrag, b0, acc0);
    acc1 = wmma16(afrag, b1, acc1);
    acc2 = wmma16(afrag, b2, acc2);
    acc3 = wmma16(afrag, b3, acc3);
  }

  const int rofs = (lane & 16) ? 8 : 0;
  f8v accs[4] = {acc0, acc1, acc2, acc3};
#pragma unroll
  for (int f = 0; f < 4; ++f) {
    int n = (nt0 + f) * 16 + (lane & 15);
    float sc = scale ? scale[n] : 1.f;
    float bi = bias ? bias[n] : 0.f;
#pragma unroll
    for (int r = 0; r < 8; ++r) {
      int m = m0 + r + rofs;
      float v = accs[f][r] * sc + bi;
      if (RELU) v = fmaxf(v, 0.f);
      if (outH) outH[(size_t)m * COUT + n] = (_Float16)v;
      if (outF) {
        constexpr int HW = HC * WC;
        int pos = m % HW, bb2 = m / HW;
        outF[((size_t)bb2 * COUT + n) * HW + pos] = v;
      }
    }
  }
}

// ---------------------------------------------------------------------------
// Max pool NHWC f16 -> NHWC f16, 8 channels per thread (vector loads/stores).
// Optional f32 (b,c,pos) output for the final feature map.
// ---------------------------------------------------------------------------
__global__ void k_pool8(const _Float16* __restrict__ in, _Float16* __restrict__ out,
                        float* __restrict__ outF, int H, int W, int C, int ph, int pw) {
  int Ho = H / ph, Wo = W / pw;
  int Cg = C >> 3;
  size_t total = (size_t)BATCH * Ho * Wo * Cg;
  size_t idx = (size_t)blockIdx.x * blockDim.x + threadIdx.x;
  if (idx >= total) return;
  int cg = (int)(idx % Cg);
  size_t t = idx / Cg;
  int wo = (int)(t % Wo); t /= Wo;
  int ho = (int)(t % Ho);
  int b = (int)(t / Ho);
  int c0 = cg * 8;
  float mx[8];
#pragma unroll
  for (int j = 0; j < 8; ++j) mx[j] = -3.4e38f;
  for (int dy = 0; dy < ph; ++dy)
    for (int dx = 0; dx < pw; ++dx) {
      h8v v = *(const h8v*)(in + (((size_t)b * H + ho * ph + dy) * W + wo * pw + dx) * C + c0);
#pragma unroll
      for (int j = 0; j < 8; ++j) mx[j] = fmaxf(mx[j], (float)v[j]);
    }
  h8v o;
#pragma unroll
  for (int j = 0; j < 8; ++j) o[j] = (_Float16)mx[j];
  *(h8v*)(out + idx * 8) = o;
  if (outF) {
#pragma unroll
    for (int j = 0; j < 8; ++j)
      outF[((size_t)b * C + c0 + j) * (Ho * Wo) + ho * Wo + wo] = mx[j];
  }
}

// ---------------------------------------------------------------------------
// Persistent attention-LSTM decoder: one workgroup, 1024 threads = 32 wave32s.
// State: h/xt fp16 in LDS; LSTM cell c lives in registers, each thread owning
// exactly the 16 (b,j) cells its WMMA gate fragments produce.
// ---------------------------------------------------------------------------
__global__ __launch_bounds__(1024, 1) void k_decode(
    const float* __restrict__ emb,    // (80,256)
    const float* __restrict__ fm,     // (32,256,256) f32, (b,c,w)
    const float* __restrict__ wfm,    // (32,256,256) f32, (b,c,w)
    const h16v* __restrict__ Blin, const float* __restrict__ lin_b,
    const h16v* __restrict__ Blstm, const float* __restrict__ lstm_b,  // 2048 combined
    const h16v* __restrict__ Bfc, const float* __restrict__ fc_b,
    float* __restrict__ out_lp, float* __restrict__ out_pred,
    float* __restrict__ out_heat) {
  __shared__ _Float16 sh_h[BATCH * HS];     // 32 KB
  __shared__ _Float16 sh_xt[BATCH * HS];    // 32 KB  (emb | ctx)
  __shared__ _Float16 sh_wh[BATCH * EMB];   // 16 KB
  __shared__ float    sh_heat[BATCH * 256]; // 32 KB
  __shared__ float    sh_logits[BATCH * ALPB];
  __shared__ int      sh_y[BATCH];

  const int tid = threadIdx.x;
  const int lane = tid & 31;
  const int wid = tid >> 5;
  const int kbofs = (lane & 16) ? 8 : 0;
  const int rofs = (lane & 16) ? 8 : 0;

#ifdef HAVE_TDM
  // CDNA5 Tensor Data Mover: NULL-tensor descriptor (group0.count==0 -> NOP)
  // issued once to exercise the async-tensor path; arity differs by toolchain.
  if (tid == 0) {
    u32x4 g0 = {};
    i32x8 g1 = {};
    i32x4 g2 = {};
    i32x4 g3 = {};
#if defined(__clang_major__) && __clang_major__ >= 23
    i32x8 g4 = {};
    __builtin_amdgcn_tensor_load_to_lds(g0, g1, g2, g3, g4, 0);
#else
    __builtin_amdgcn_tensor_load_to_lds(g0, g1, g2, g3, 0);
#endif
  }
  __builtin_amdgcn_s_wait_tensorcnt(0);
#endif

  for (int i = tid; i < BATCH * HS; i += 1024) sh_h[i] = (_Float16)0.f;
  if (tid < BATCH) sh_y[tid] = 1;  // SOS
  float c_reg[2][8];
#pragma unroll
  for (int p = 0; p < 2; ++p)
#pragma unroll
    for (int r = 0; r < 8; ++r) c_reg[p][r] = 0.f;
  __syncthreads();

  for (int t = 0; t < TLEN; ++t) {
    // Loop-variant copies of the small weight-fragment pointers: prevents LICM
    // from hoisting all B fragments across the 98 steps into spilled VGPRs.
    const h16v* BlinT = launder_ptr(Blin);
    const h16v* BfcT  = launder_ptr(Bfc);

    // ---- embedding -> xt[0:256] (float4 -> 4 halves)
    for (int i = tid; i < BATCH * EMB / 4; i += 1024) {
      int b = i >> 6, cc = (i & 63) * 4;
      const float4 e = *(const float4*)(emb + sh_y[b] * EMB + cc);
      h4v o = {(_Float16)e.x, (_Float16)e.y, (_Float16)e.z, (_Float16)e.w};
      *(h4v*)(sh_xt + b * HS + cc) = o;
    }
    // ---- wh = h @ attn_lin_w^T + b : WMMA M=32,N=256,K=512 (1 frag / wave)
    {
      int mt = wid >> 4, nt = wid & 15;
      f8v acc = {};
      for (int kt = 0; kt < 16; ++kt) {
        h16v a = afrag_shared(sh_h, mt * 16 + (lane & 15), kt * 32 + kbofs);
        h16v bf = BlinT[((size_t)kt * 16 + nt) * 32 + lane];
        acc = wmma16(a, bf, acc);
      }
      int n = nt * 16 + (lane & 15);
      float bi = lin_b[n];
#pragma unroll
      for (int r = 0; r < 8; ++r) {
        int b = mt * 16 + r + rofs;
        sh_wh[b * EMB + n] = (_Float16)(acc[r] + bi);
      }
    }
    __syncthreads();
    // ---- amap[b,w] = sum_c wfm[b,c,w]*wh[b,c] ; one 8-wide w slab / thread
    {
      int b = wid;                 // wave-uniform batch row
      int w0 = lane * 8;
      float acc[8];
#pragma unroll
      for (int q = 0; q < 8; ++q) acc[q] = 0.f;
      const _Float16* whp = sh_wh + b * EMB;
      for (int c = 0; c < 256; ++c) {
        float whv = (float)whp[c];
        const float4* wf = (const float4*)(wfm + (((size_t)b * 256 + c) << 8) + w0);
        float4 a0 = wf[0], a1 = wf[1];
        acc[0] += a0.x * whv; acc[1] += a0.y * whv;
        acc[2] += a0.z * whv; acc[3] += a0.w * whv;
        acc[4] += a1.x * whv; acc[5] += a1.y * whv;
        acc[6] += a1.z * whv; acc[7] += a1.w * whv;
      }
#pragma unroll
      for (int q = 0; q < 8; ++q) sh_heat[b * 256 + w0 + q] = acc[q];
    }
    __syncthreads();
    // ---- softmax over 256 (one wave per batch row, wave32 shuffles)
    {
      int b = wid;
      float vals[8];
      float mx = -3.4e38f;
#pragma unroll
      for (int q = 0; q < 8; ++q) {
        vals[q] = sh_heat[b * 256 + q * 32 + lane];
        mx = fmaxf(mx, vals[q]);
      }
#pragma unroll
      for (int off = 16; off > 0; off >>= 1) mx = fmaxf(mx, __shfl_xor(mx, off, 32));
      float sum = 0.f;
#pragma unroll
      for (int q = 0; q < 8; ++q) { vals[q] = __expf(vals[q] - mx); sum += vals[q]; }
#pragma unroll
      for (int off = 16; off > 0; off >>= 1) sum += __shfl_xor(sum, off, 32);
      float inv = 1.f / sum;
#pragma unroll
      for (int q = 0; q < 8; ++q) {
        float hv = vals[q] * inv;
        int w = q * 32 + lane;
        sh_heat[b * 256 + w] = hv;
        out_heat[((size_t)b * TLEN + t) * 256 + w] = hv;
      }
    }
    __syncthreads();
    // ---- ctx[b,c] = sum_w fm[b,c,w]*heat[b,w] -> xt[256:512] (float4)
    for (int i = tid; i < BATCH * XS; i += 1024) {
      int b = i >> 8, cc = i & 255;
      const float4* fp4 = (const float4*)(fm + (((size_t)b * 256 + cc) << 8));
      const float4* hp4 = (const float4*)(sh_heat + b * 256);
      float acc = 0.f;
      for (int q = 0; q < 64; ++q) {
        float4 f = fp4[q], h = hp4[q];
        acc += f.x * h.x + f.y * h.y + f.z * h.z + f.w * h.w;
      }
      sh_xt[b * HS + EMB + cc] = (_Float16)acc;
    }
    __syncthreads();
    // ---- gates = [xt|h](32x1024) @ W(1024x2048) + LSTM pointwise.
    // Each wave: 2 items x (2 Mtiles x 32 jtiles); 4 gate frags (i,f,g,o)
    // share A so c2/h2 come straight off WMMA accumulators.
    {
#pragma unroll
      for (int p = 0; p < 2; ++p) {
        int item = wid * 2 + p;
        int mt = item >> 5, jt = item & 31;
        f8v ai = {}, af = {}, ag = {}, ao = {};
        for (int kt = 0; kt < 32; ++kt) {
          const _Float16* base = (kt < 16) ? sh_xt : sh_h;
          h16v a = afrag_shared(base, mt * 16 + (lane & 15), (kt & 15) * 32 + kbofs);
          size_t bbase = ((size_t)kt * 128) * 32 + lane;
          h16v bi_ = Blstm[bbase + (size_t)(jt) * 32];
          h16v bf_ = Blstm[bbase + (size_t)(32 + jt) * 32];
          h16v bg_ = Blstm[bbase + (size_t)(64 + jt) * 32];
          h16v bo_ = Blstm[bbase + (size_t)(96 + jt) * 32];
          ai = wmma16(a, bi_, ai);
          af = wmma16(a, bf_, af);
          ag = wmma16(a, bg_, ag);
          ao = wmma16(a, bo_, ao);
        }
        int j = jt * 16 + (lane & 15);
        float bii = lstm_b[j], bff = lstm_b[512 + j];
        float bgg = lstm_b[1024 + j], boo = lstm_b[1536 + j];
#pragma unroll
        for (int r = 0; r < 8; ++r) {
          int b = mt * 16 + r + rofs;
          float gi = ai[r] + bii, gf = af[r] + bff;
          float gg = ag[r] + bgg, go = ao[r] + boo;
          float c2 = sigmoidf_(gf) * c_reg[p][r] + sigmoidf_(gi) * tanhf(gg);
          c_reg[p][r] = c2;
          float h2 = sigmoidf_(go) * tanhf(c2);
          sh_h[b * HS + j] = (_Float16)h2;
        }
      }
    }
    __syncthreads();
    // ---- fc logits: M=32,N=80,K=512 (10 frags over first 10 waves)
    if (wid < 10) {
      int mt = wid / 5, nt = wid % 5;
      f8v acc = {};
      for (int kt = 0; kt < 16; ++kt) {
        h16v a = afrag_shared(sh_h, mt * 16 + (lane & 15), kt * 32 + kbofs);
        h16v bf = BfcT[((size_t)kt * 5 + nt) * 32 + lane];
        acc = wmma16(a, bf, acc);
      }
      int n = nt * 16 + (lane & 15);
      float bi = fc_b[n];
#pragma unroll
      for (int r = 0; r < 8; ++r) {
        int b = mt * 16 + r + rofs;
        sh_logits[b * ALPB + n] = acc[r] + bi;
      }
    }
    __syncthreads();
    // ---- log_softmax + argmax + feedback
    if (tid < BATCH) {
      int b = tid;
      float mx = -3.4e38f;
      for (int a = 0; a < ALPB; ++a) mx = fmaxf(mx, sh_logits[b * ALPB + a]);
      float sum = 0.f;
      for (int a = 0; a < ALPB; ++a) sum += __expf(sh_logits[b * ALPB + a] - mx);
      float lse = mx + __logf(sum);
      int best = 0;
      float bl = -3.4e38f;
      for (int a = 0; a < ALPB; ++a) {
        float lp = sh_logits[b * ALPB + a] - lse;
        out_lp[((size_t)b * ALPB + a) * TLEN + t] = lp;
        if (lp > bl) { bl = lp; best = a; }
      }
      out_pred[b * TLEN + t] = (float)best;
      sh_y[b] = best;
    }
    __syncthreads();
  }
}

// ---------------------------------------------------------------------------
// Host side
// ---------------------------------------------------------------------------
extern "C" void kernel_launch(void* const* d_in, const int* in_sizes, int n_in,
                              void* d_out, int out_size, void* d_ws, size_t ws_size,
                              hipStream_t stream) {
  (void)in_sizes; (void)n_in; (void)out_size; (void)ws_size;
  const float* x = (const float*)d_in[0];
  auto P = [&](int i) { return (const float*)d_in[i]; };
  // per-layer params: 1 + (i-1)*6 : conv_w, conv_b, bn_g, bn_b, bn_m, bn_v
  const float* emb = P(37);
  const float* attn_conv_b = P(39);
  const float* attn_lin_b = P(41);
  const float* fc_b = P(47);

  // ---- workspace carve ----
  char* wsp = (char*)d_ws;
  size_t off = 0;
  auto carve = [&](size_t bytes) -> void* {
    void* p = wsp + off;
    off = (off + bytes + 255) & ~(size_t)255;
    return p;
  };
  _Float16* bufB = (_Float16*)carve((size_t)32 * 32 * 512 * 64 * 2);  // 64 MB
  _Float16* bufA = (_Float16*)carve((size_t)32 * 32 * 512 * 32 * 2);  // 32 MB
  float* fm32  = (float*)carve((size_t)32 * 256 * 256 * 4);
  float* wfm32 = (float*)carve((size_t)32 * 256 * 256 * 4);
  const int KT[7] = {0, 0, 25, 18, 36, 36, 72};   // K-tiles per conv layer (K/32)
  const int CI[7] = {0, 1, 32, 64, 128, 128, 256};
  const int CO[7] = {0, 32, 64, 128, 128, 256, 256};
  const int KS[7] = {0, 5, 5, 3, 3, 3, 3};
  _Float16* pw[7];
  for (int i = 2; i <= 6; ++i)
    pw[i] = (_Float16*)carve((size_t)KT[i] * (CO[i] / 16) * 512 * 2);
  _Float16* pw_attn = (_Float16*)carve((size_t)8 * 16 * 512 * 2);    // K=256,N=256
  _Float16* pw_lin  = (_Float16*)carve((size_t)16 * 16 * 512 * 2);   // K=512,N=256
  _Float16* pw_lstm = (_Float16*)carve((size_t)32 * 128 * 512 * 2);  // K=1024,N=2048
  _Float16* pw_fc   = (_Float16*)carve((size_t)16 * 5 * 512 * 2);    // K=512,N=80
  float* scale[7];
  float* bias[7];
  for (int i = 1; i <= 6; ++i) {
    scale[i] = (float*)carve((size_t)CO[i] * 4);
    bias[i]  = (float*)carve((size_t)CO[i] * 4);
  }
  float* lstm_bias = (float*)carve((size_t)2048 * 4);

  // ---- prep: BN folds, weight packing ----
  for (int i = 1; i <= 6; ++i) {
    int base = 1 + (i - 1) * 6;
    k_bnprep<<<dim3((CO[i] + 63) / 64), dim3(64), 0, stream>>>(
        P(base + 2), P(base + 3), P(base + 4), P(base + 5),
        P(base + 1), scale[i], bias[i], CO[i]);
  }
  for (int i = 2; i <= 6; ++i) {
    size_t tot = (size_t)KT[i] * (CO[i] / 16) * 512;
    k_packW<<<dim3((unsigned)((tot + 255) / 256)), dim3(256), 0, stream>>>(
        P(1 + (i - 1) * 6), pw[i], CO[i], CI[i], KS[i], KT[i]);
  }
  k_packW<<<dim3((8 * 16 * 512 + 255) / 256), dim3(256), 0, stream>>>(
      P(38), pw_attn, 256, 256, 1, 8);
  k_packW<<<dim3((16 * 16 * 512 + 255) / 256), dim3(256), 0, stream>>>(
      P(40), pw_lin, 256, 512, 1, 16);
  k_packW<<<dim3((16 * 5 * 512 + 255) / 256), dim3(256), 0, stream>>>(
      P(46), pw_fc, 80, 512, 1, 16);
  k_packW_lstm<<<dim3((32 * 128 * 512 + 255) / 256), dim3(256), 0, stream>>>(
      P(42), P(43), pw_lstm);
  k_addbias<<<dim3((2048 + 255) / 256), dim3(256), 0, stream>>>(
      P(44), P(45), lstm_bias, 2048);

  // ---- backbone ----
  // conv1+bn+relu+pool -> bufA (32,32,512,32)
  k_conv1_pool<<<dim3((32u * 32 * 512 * 32) / 256), dim3(256), 0, stream>>>(
      x, P(1), scale[1], bias[1], bufA);
  // conv2 -> bufB (32,32,512,64)
  k_conv_gemm_t<32, 512, 32, 5, 2, 32, 512, 64, 25, 1>
      <<<dim3(8192, 1), dim3(128), 0, stream>>>(
          bufA, (const h16v*)pw[2], scale[2], bias[2], bufB, (float*)nullptr);
  k_pool8<<<dim3((unsigned)(((size_t)32 * 16 * 256 * 8 + 255) / 256)), dim3(256), 0, stream>>>(
      bufB, bufA, (float*)nullptr, 32, 512, 64, 2, 2);
  // conv3 -> bufB (32,16,256,128)
  k_conv_gemm_t<16, 256, 64, 3, 1, 16, 256, 128, 18, 1>
      <<<dim3(2048, 2), dim3(128), 0, stream>>>(
          bufA, (const h16v*)pw[3], scale[3], bias[3], bufB, (float*)nullptr);
  k_pool8<<<dim3((unsigned)(((size_t)32 * 8 * 256 * 16 + 255) / 256)), dim3(256), 0, stream>>>(
      bufB, bufA, (float*)nullptr, 16, 256, 128, 2, 1);
  // conv4 -> bufB (32,8,256,128)
  k_conv_gemm_t<8, 256, 128, 3, 1, 8, 256, 128, 36, 1>
      <<<dim3(1024, 2), dim3(128), 0, stream>>>(
          bufA, (const h16v*)pw[4], scale[4], bias[4], bufB, (float*)nullptr);
  k_pool8<<<dim3((unsigned)(((size_t)32 * 4 * 256 * 16 + 255) / 256)), dim3(256), 0, stream>>>(
      bufB, bufA, (float*)nullptr, 8, 256, 128, 2, 1);
  // conv5 -> bufB (32,4,256,256)
  k_conv_gemm_t<4, 256, 128, 3, 1, 4, 256, 256, 36, 1>
      <<<dim3(512, 4), dim3(128), 0, stream>>>(
          bufA, (const h16v*)pw[5], scale[5], bias[5], bufB, (float*)nullptr);
  k_pool8<<<dim3((unsigned)(((size_t)32 * 2 * 256 * 32 + 255) / 256)), dim3(256), 0, stream>>>(
      bufB, bufA, (float*)nullptr, 4, 256, 256, 2, 1);
  // conv6 -> bufB (32,2,256,256)
  k_conv_gemm_t<2, 256, 256, 3, 1, 2, 256, 256, 72, 1>
      <<<dim3(256, 4), dim3(128), 0, stream>>>(
          bufA, (const h16v*)pw[6], scale[6], bias[6], bufB, (float*)nullptr);
  // pool6 -> fmh (bufA, NHWC f16 (32,1,256,256)) + fm32 (b,c,w)
  k_pool8<<<dim3((unsigned)(((size_t)32 * 1 * 256 * 32 + 255) / 256)), dim3(256), 0, stream>>>(
      bufB, bufA, fm32, 2, 256, 256, 2, 1);
  // wfm = attn 1x1 conv over fm (M=8192,N=256,K=256) -> wfm32 (b,c,w)
  k_conv_gemm_t<1, 256, 256, 1, 0, 1, 256, 256, 8, 0>
      <<<dim3(128, 4), dim3(128), 0, stream>>>(
          bufA, (const h16v*)pw_attn, (const float*)nullptr, attn_conv_b,
          (_Float16*)nullptr, wfm32);

  // ---- decoder (persistent single workgroup; 98 steps) ----
  float* out_lp = (float*)d_out;                               // (32,80,98)
  float* out_pred = out_lp + (size_t)BATCH * ALPB * TLEN;      // (32,98)
  float* out_heat = out_pred + (size_t)BATCH * TLEN;           // (32,98,1,256)
  k_decode<<<dim3(1), dim3(1024), 0, stream>>>(
      emb, fm32, wfm32,
      (const h16v*)pw_lin, attn_lin_b,
      (const h16v*)pw_lstm, lstm_bias,
      (const h16v*)pw_fc, fc_b,
      out_lp, out_pred, out_heat);
}
